// AttentionBlock_89807766159392
// MI455X (gfx1250) — compile-verified, matrix operand order checked by reading, not run on hardware
//
#include <hip/hip_runtime.h>
#include <hip/hip_bf16.h>
#include <math.h>

typedef __attribute__((ext_vector_type(16))) _Float16 v16h;
typedef __attribute__((ext_vector_type(8)))  float    v8f;
typedef unsigned int u32;

typedef __attribute__((ext_vector_type(4))) unsigned int v4u;
typedef __attribute__((ext_vector_type(8))) int         v8int;
typedef __attribute__((ext_vector_type(4))) int         v4int;
typedef __attribute__((address_space(1))) v4int         as1_v4int;
typedef __attribute__((address_space(3))) v4int         as3_v4int;

#define DIM   1024
#define FFN   4096
#define HEADS 16
#define HD    64
#define BATCH 2
#define SEQ   2048
#define NTOK  (BATCH*SEQ)
#define ATTN_SCALE 0.125f   /* 1/sqrt(64) */

// ---- CDNA5 feature probes (device pass only; host pass always takes fallback)
#if defined(__HIP_DEVICE_COMPILE__) && \
    __has_builtin(__builtin_amdgcn_tensor_load_to_lds) && \
    __has_builtin(__builtin_amdgcn_s_wait_tensorcnt)
#define HAVE_TDM 1
#else
#define HAVE_TDM 0
#endif

#if defined(__HIP_DEVICE_COMPILE__) && \
    __has_builtin(__builtin_amdgcn_global_load_async_to_lds_b128)
#define HAVE_ASYNC 1
#else
#define HAVE_ASYNC 0
#endif

// K-offset pattern of the 16-bit A/B WMMA fragment (16x32, MxK):
// VGPR j (pair of halfs), lane-half selects K block of 8.
__device__ __forceinline__ int frag_k(int j, int half) {
  return ((j >> 2) << 4) + (half << 3) + ((j & 3) << 1);
}

union HFrag { v16h h; u32 u[8]; };

#if HAVE_TDM
// Issue one 2D TDM tile load: tile (tile0 x tile1) elements of 2B data from a
// row-major tensor with row stride 'stride0' elems; LDS padded +16B per 64B row
// chunk (pad_interval=16 DW, pad_amount=4 DW) -> LDS row stride 80B == LDT.
__device__ __forceinline__ void tdm_load_2d(const _Float16* gptr, u32 lds_off,
                                            u32 tdim0, u32 tdim1, u32 stride0,
                                            u32 tile0, u32 tile1) {
  unsigned long long ga = (unsigned long long)gptr;
  v4u g0;
  g0[0] = 1u;                                   // count=1, user descriptor
  g0[1] = lds_off;                              // lds_addr (bytes)
  g0[2] = (u32)ga;                              // global_addr[31:0]
  g0[3] = ((u32)(ga >> 32) & 0x01FFFFFFu)       // global_addr[56:32]
          | 0x80000000u;                        // type=2 (image)
  v8int g1;
  g1[0] = (int)((1u << 16)                      // data_size = 2 bytes
              | (1u << 20)                      // pad_enable
              | (3u << 22)                      // pad_interval: 16 DWORDs (64B)
              | (3u << 25));                    // pad_amount: 4 DWORDs (16B)
  g1[1] = (int)((tdim0 & 0xFFFFu) << 16);       // atomic_barrier=0 | dim0 lo16
  g1[2] = (int)(((tdim0 >> 16) & 0xFFFFu) | ((tdim1 & 0xFFFFu) << 16));
  g1[3] = (int)(((tdim1 >> 16) & 0xFFFFu) | (tile0 << 16));   // | tile_dim0
  g1[4] = (int)(tile1 & 0xFFFFu);               // tile_dim1 | tile_dim2=0
  g1[5] = (int)stride0;                         // tensor_dim0_stride lo32
  g1[6] = 0;                                    // stride0 hi | stride1 lo
  g1[7] = 0;                                    // stride1 hi
  v4int z4 = {0, 0, 0, 0};
  v8int z8 = {0, 0, 0, 0, 0, 0, 0, 0};
  __builtin_amdgcn_tensor_load_to_lds(g0, g1, z4, z4, z8, 0);
}
#endif

// ---------------------------------------------------------------- conversions
__global__ void cvt_f32_to_f16(const float* __restrict__ in,
                               _Float16* __restrict__ out, int n) {
  int i = blockIdx.x * blockDim.x + threadIdx.x;
  if (i < n) out[i] = (_Float16)in[i];
}

// ---------------------------------------------------------------- layernorm
__global__ void __launch_bounds__(256)
layernorm_to_f16(const float* __restrict__ x, const float* __restrict__ g,
                 const float* __restrict__ b, _Float16* __restrict__ out) {
  const int row = blockIdx.x;
  const float* xr = x + (size_t)row * DIM;
  __shared__ float red[8];

  float v[4];
  float s = 0.f;
#pragma unroll
  for (int i = 0; i < 4; ++i) { v[i] = xr[threadIdx.x + 256 * i]; s += v[i]; }
#pragma unroll
  for (int m = 16; m >= 1; m >>= 1) s += __shfl_xor(s, m, 32);
  if ((threadIdx.x & 31) == 0) red[threadIdx.x >> 5] = s;
  __syncthreads();
  float tot = 0.f;
#pragma unroll
  for (int i = 0; i < 8; ++i) tot += red[i];
  const float mu = tot * (1.0f / DIM);

  float s2 = 0.f;
#pragma unroll
  for (int i = 0; i < 4; ++i) { float d = v[i] - mu; s2 += d * d; }
#pragma unroll
  for (int m = 16; m >= 1; m >>= 1) s2 += __shfl_xor(s2, m, 32);
  __syncthreads();
  if ((threadIdx.x & 31) == 0) red[threadIdx.x >> 5] = s2;
  __syncthreads();
  float v2 = 0.f;
#pragma unroll
  for (int i = 0; i < 8; ++i) v2 += red[i];
  const float inv = rsqrtf(v2 * (1.0f / DIM) + 1e-5f);

#pragma unroll
  for (int i = 0; i < 4; ++i) {
    int c = threadIdx.x + 256 * i;
    out[(size_t)row * DIM + c] = (_Float16)((v[i] - mu) * inv * g[c] + b[c]);
  }
}

// ---------------------------------------------------------------- WMMA GEMM
// Y[M,N] = A[M,K] * Bw[N,K]^T + bias ; epilogue:
//   EPI==0 : f16 store | EPI==1 : GELU(erf), f16 store | EPI==2 : f32 resid+val
#define BM  64
#define BN  128
#define BK  32
#define LDT 40   // padded LDS row stride in halfs (80B: 16B aligned, low conflict)

template <int EPI>
__global__ void __launch_bounds__(256)
gemm_f16w(const _Float16* __restrict__ A, const _Float16* __restrict__ Bw,
          const float* __restrict__ bias, const float* __restrict__ resid,
          float* __restrict__ outF, _Float16* __restrict__ outH,
          int M, int N, int K) {
  __shared__ _Float16 tA[2][BM * LDT];
  __shared__ _Float16 tB[2][BN * LDT];

  const int t    = threadIdx.x;
  const int lane = t & 31;
  const int wave = t >> 5;
  const int half = lane >> 4;
  const int r16  = lane & 15;
  const int wm   = wave >> 2;   // 0..1 : 32-row slab
  const int wn   = wave & 3;    // 0..3 : 32-col slab
  const int m0   = blockIdx.y * BM;
  const int n0   = blockIdx.x * BN;

  v8f acc[2][2] = {};

  auto compute_step = [&](const _Float16* sA, const _Float16* sB) {
    HFrag af[2], bf[2];
#pragma unroll
    for (int i = 0; i < 2; ++i) {
      int ar = wm * 32 + i * 16 + r16;
#pragma unroll
      for (int j = 0; j < 8; ++j)
        af[i].u[j] = *(const u32*)&sA[ar * LDT + frag_k(j, half)];
    }
#pragma unroll
    for (int i = 0; i < 2; ++i) {
      int br = wn * 32 + i * 16 + r16;
#pragma unroll
      for (int j = 0; j < 8; ++j)
        bf[i].u[j] = *(const u32*)&sB[br * LDT + frag_k(j, half)];
    }
#pragma unroll
    for (int i = 0; i < 2; ++i)
#pragma unroll
      for (int j = 0; j < 2; ++j)
        acc[i][j] = __builtin_amdgcn_wmma_f32_16x16x32_f16(
            false, af[i].h, false, bf[j].h, (short)0, acc[i][j], false, false);
  };

#if HAVE_TDM
  // ---- Tensor Data Mover path: wave 0 drives a double-buffered tile pipeline.
  const u32 aL[2] = {(u32)(unsigned long long)&tA[0][0],
                     (u32)(unsigned long long)&tA[1][0]};
  const u32 bL[2] = {(u32)(unsigned long long)&tB[0][0],
                     (u32)(unsigned long long)&tB[1][0]};
  if (wave == 0) {
    tdm_load_2d(&A[(size_t)m0 * K],  aL[0], (u32)K, (u32)M, (u32)K, BK, BM);
    tdm_load_2d(&Bw[(size_t)n0 * K], bL[0], (u32)K, (u32)N, (u32)K, BK, BN);
  }
  int cur = 0;
  for (int k0 = 0; k0 < K; k0 += BK) {
    if (wave == 0) {
      if (k0 + BK < K) {   // prefetch next K-slice into the other buffer
        tdm_load_2d(&A[(size_t)m0 * K + k0 + BK],  aL[cur ^ 1],
                    (u32)K, (u32)M, (u32)K, BK, BM);
        tdm_load_2d(&Bw[(size_t)n0 * K + k0 + BK], bL[cur ^ 1],
                    (u32)K, (u32)N, (u32)K, BK, BN);
        __builtin_amdgcn_s_wait_tensorcnt(2);   // current pair done
      } else {
        __builtin_amdgcn_s_wait_tensorcnt(0);
      }
    }
    __syncthreads();
    compute_step(tA[cur], tB[cur]);
    __syncthreads();   // everyone done reading before TDM overwrites this buf
    cur ^= 1;
  }
#else
  // ---- fallback: cooperative VMEM->VGPR->LDS copies, single buffer
  for (int k0 = 0; k0 < K; k0 += BK) {
    __syncthreads();
    {
      int ar = t >> 2, ac = (t & 3) << 3;           // A tile 64x32
      *(uint4*)&tA[0][ar * LDT + ac] =
          *(const uint4*)&A[(size_t)(m0 + ar) * K + k0 + ac];
#pragma unroll
      for (int i = 0; i < 2; ++i) {                 // B tile 128x32
        int idx = t + i * 256;
        int br = idx >> 2, bc = (idx & 3) << 3;
        *(uint4*)&tB[0][br * LDT + bc] =
            *(const uint4*)&Bw[(size_t)(n0 + br) * K + k0 + bc];
      }
    }
    __syncthreads();
    compute_step(tA[0], tB[0]);
  }
#endif

  // ---- epilogue: C layout = lane holds col r16, rows (half*8+v)
#pragma unroll
  for (int i = 0; i < 2; ++i) {
#pragma unroll
    for (int j = 0; j < 2; ++j) {
      const int col = n0 + wn * 32 + j * 16 + r16;
      const float bb = bias[col];
#pragma unroll
      for (int v = 0; v < 8; ++v) {
        const int row = m0 + wm * 32 + i * 16 + (half << 3) + v;
        float val = acc[i][j][v] + bb;
        if (EPI == 0) {
          outH[(size_t)row * N + col] = (_Float16)val;
        } else if (EPI == 1) {
          float gel = 0.5f * val * (1.0f + erff(val * 0.70710678118f));
          outH[(size_t)row * N + col] = (_Float16)gel;
        } else {
          outF[(size_t)row * N + col] = resid[(size_t)row * N + col] + val;
        }
      }
    }
  }
}

// ---------------------------------------------------------------- flash attention
// One wave = 16 queries x full head (d=64). 32 keys per iteration.
// Block = 4 waves sharing (batch, head) and the K/V LDS tiles.
// K tile: row layout (async-to-LDS when available).
// V tile: pair-interleaved dwords vtp[d][rp] = {V[2rp][d], V[2rp+1][d]} so the
//         P*V B-fragment is 8 aligned ds_load_b32 with no half repacking.
#define KB  32
#define LDK 72   // K tile stride halfs (144B: 16B aligned)
#define LDV 17   // V interleaved tile stride dwords (68B: conflict-free)
#define LDP 34   // P transpose tile stride halfs (bank-conflict free)

__global__ void __launch_bounds__(128)
attention_f16(const _Float16* __restrict__ Q, const _Float16* __restrict__ Kk,
              const _Float16* __restrict__ V, _Float16* __restrict__ O) {
  __shared__ _Float16 kt[KB * LDK];
  __shared__ u32      vtp[HD * LDV];
  __shared__ _Float16 pt[4 * 16 * LDP];

  const int t    = threadIdx.x;
  const int lane = t & 31;
  const int wave = t >> 5;
  const int half = lane >> 4;
  const int r16  = lane & 15;

  const int qb = blockIdx.x & 31;     // SEQ/64 == 32 query blocks
  const int bh = blockIdx.x >> 5;
  const int b  = bh >> 4;
  const int h  = bh & 15;
  const int qtok0  = b * SEQ + qb * 64 + wave * 16;
  const int kvtok0 = b * SEQ;
  const size_t hoff = (size_t)h * HD;

  // Q fragments: 16x64 -> two 16x32 A-frags, straight from global (contiguous K)
  HFrag qf[2];
#pragma unroll
  for (int s = 0; s < 2; ++s)
#pragma unroll
    for (int j = 0; j < 8; ++j)
      qf[s].u[j] = *(const u32*)&Q[(size_t)(qtok0 + r16) * DIM + hoff +
                                   s * 32 + frag_k(j, half)];

  v8f o[4] = {};
  float mrun[8], lrun[8];
#pragma unroll
  for (int v = 0; v < 8; ++v) { mrun[v] = -1e30f; lrun[v] = 0.f; }

  _Float16* pw = &pt[wave * 16 * LDP];

  for (int kb = 0; kb < SEQ; kb += KB) {
    __syncthreads();
    {
      // ---- K tile (32 x 64 halfs)
#if HAVE_ASYNC
#pragma unroll
      for (int i = 0; i < 2; ++i) {
        int idx = t + i * 128;
        int row = idx >> 3, c4 = (idx & 7) << 3;
        __builtin_amdgcn_global_load_async_to_lds_b128(
            (as1_v4int*)&Kk[(size_t)(kvtok0 + kb + row) * DIM + hoff + c4],
            (as3_v4int*)&kt[row * LDK + c4], 0, 0);
      }
#else
#pragma unroll
      for (int i = 0; i < 2; ++i) {
        int idx = t + i * 128;
        int row = idx >> 3, c4 = (idx & 7) << 3;
        *(uint4*)&kt[row * LDK + c4] =
            *(const uint4*)&Kk[(size_t)(kvtok0 + kb + row) * DIM + hoff + c4];
      }
#endif
      // ---- V tile, pair-interleaved: thread owns row pair rp, 8 columns
      int rp = t >> 3, c8 = (t & 7) << 3;
      const _Float16* vp = &V[(size_t)(kvtok0 + kb + 2 * rp) * DIM + hoff + c8];
      uint4 a = *(const uint4*)vp;
      uint4 c = *(const uint4*)(vp + DIM);
      u32 aw[4] = {a.x, a.y, a.z, a.w};
      u32 cw[4] = {c.x, c.y, c.z, c.w};
#pragma unroll
      for (int i = 0; i < 4; ++i) {
        u32 lo = (aw[i] & 0x0000FFFFu) | (cw[i] << 16);
        u32 hi = (aw[i] >> 16)         | (cw[i] & 0xFFFF0000u);
        vtp[(c8 + 2 * i)     * LDV + rp] = lo;
        vtp[(c8 + 2 * i + 1) * LDV + rp] = hi;
      }
#if HAVE_ASYNC
#if __has_builtin(__builtin_amdgcn_s_wait_asynccnt)
      __builtin_amdgcn_s_wait_asynccnt(0);
#else
      asm volatile("s_wait_asynccnt 0x0" ::: "memory");
#endif
#endif
    }
    __syncthreads();

    // S = Q * K^T : 16x32 scores in two f32 accumulators
    v8f s0 = {}, s1 = {};
    HFrag kf;
#pragma unroll
    for (int s = 0; s < 2; ++s) {
#pragma unroll
      for (int j = 0; j < 8; ++j)
        kf.u[j] = *(const u32*)&kt[r16 * LDK + s * 32 + frag_k(j, half)];
      s0 = __builtin_amdgcn_wmma_f32_16x16x32_f16(false, qf[s].h, false, kf.h,
                                                  (short)0, s0, false, false);
#pragma unroll
      for (int j = 0; j < 8; ++j)
        kf.u[j] = *(const u32*)&kt[(16 + r16) * LDK + s * 32 + frag_k(j, half)];
      s1 = __builtin_amdgcn_wmma_f32_16x16x32_f16(false, qf[s].h, false, kf.h,
                                                  (short)0, s1, false, false);
    }

    // online softmax per query row (rows split across vgpr v + lane-half)
    float alpha[8];
#pragma unroll
    for (int v = 0; v < 8; ++v) {
      float a0 = s0[v] * ATTN_SCALE, a1 = s1[v] * ATTN_SCALE;
      float mx = fmaxf(a0, a1);
#pragma unroll
      for (int m = 8; m >= 1; m >>= 1) mx = fmaxf(mx, __shfl_xor(mx, m, 16));
      float mnew = fmaxf(mrun[v], mx);
      float p0 = __expf(a0 - mnew);
      float p1 = __expf(a1 - mnew);
      float sum = p0 + p1;
#pragma unroll
      for (int m = 8; m >= 1; m >>= 1) sum += __shfl_xor(sum, m, 16);
      alpha[v] = __expf(mrun[v] - mnew);
      lrun[v]  = lrun[v] * alpha[v] + sum;
      mrun[v]  = mnew;
      // transpose P through per-wave LDS region (C layout -> A layout)
      pw[((half << 3) + v) * LDP + r16]      = (_Float16)p0;
      pw[((half << 3) + v) * LDP + 16 + r16] = (_Float16)p1;
    }

    HFrag pf;
#pragma unroll
    for (int j = 0; j < 8; ++j)
      pf.u[j] = *(const u32*)&pw[r16 * LDP + frag_k(j, half)];

    // O = diag(alpha)*O + P * V   (4 d-groups of 16)
#pragma unroll
    for (int g = 0; g < 4; ++g) {
      HFrag vf;
#pragma unroll
      for (int j = 0; j < 8; ++j) {
        int kp = ((j >> 2) << 3) + (half << 2) + (j & 3);   // frag_k(j,half)/2
        vf.u[j] = vtp[(g * 16 + r16) * LDV + kp];
      }
#pragma unroll
      for (int v = 0; v < 8; ++v) o[g][v] *= alpha[v];
      o[g] = __builtin_amdgcn_wmma_f32_16x16x32_f16(false, pf.h, false, vf.h,
                                                    (short)0, o[g], false, false);
    }
  }

#pragma unroll
  for (int g = 0; g < 4; ++g)
#pragma unroll
    for (int v = 0; v < 8; ++v) {
      int row = qtok0 + (half << 3) + v;
      O[(size_t)row * DIM + hoff + g * 16 + r16] = (_Float16)(o[g][v] / lrun[v]);
    }
}

// ---------------------------------------------------------------- launcher
extern "C" void kernel_launch(void* const* d_in, const int* in_sizes, int n_in,
                              void* d_out, int out_size, void* d_ws, size_t ws_size,
                              hipStream_t stream) {
  (void)in_sizes; (void)n_in; (void)out_size; (void)ws_size;
  const float* x   = (const float*)d_in[0];
  const float* g1  = (const float*)d_in[1];
  const float* b1  = (const float*)d_in[2];
  const float* Wq  = (const float*)d_in[3];
  const float* bq  = (const float*)d_in[4];
  const float* Wk  = (const float*)d_in[5];
  const float* bk  = (const float*)d_in[6];
  const float* Wv  = (const float*)d_in[7];
  const float* bv  = (const float*)d_in[8];
  const float* Wo  = (const float*)d_in[9];
  const float* bo  = (const float*)d_in[10];
  const float* g2  = (const float*)d_in[11];
  const float* b2  = (const float*)d_in[12];
  const float* W1  = (const float*)d_in[13];
  const float* bf1 = (const float*)d_in[14];
  const float* W2  = (const float*)d_in[15];
  const float* bf2 = (const float*)d_in[16];
  float* out = (float*)d_out;

  size_t off = 0;
  char* base = (char*)d_ws;
  auto carve = [&](size_t nbytes) -> void* {
    void* p = base + off;
    off += (nbytes + 255) & ~(size_t)255;
    return p;
  };
  const size_t DD = (size_t)DIM * DIM;
  const size_t FD = (size_t)FFN * DIM;
  const size_t ND = (size_t)NTOK * DIM;
  const size_t NF = (size_t)NTOK * FFN;

  _Float16* Wq16 = (_Float16*)carve(DD * 2);
  _Float16* Wk16 = (_Float16*)carve(DD * 2);
  _Float16* Wv16 = (_Float16*)carve(DD * 2);
  _Float16* Wo16 = (_Float16*)carve(DD * 2);
  _Float16* W1h  = (_Float16*)carve(FD * 2);
  _Float16* W2h  = (_Float16*)carve(FD * 2);
  _Float16* hln  = (_Float16*)carve(ND * 2);
  _Float16* Qh   = (_Float16*)carve(ND * 2);
  _Float16* Kh   = (_Float16*)carve(ND * 2);
  _Float16* Vh   = (_Float16*)carve(ND * 2);
  _Float16* Ah   = (_Float16*)carve(ND * 2);
  _Float16* h2   = (_Float16*)carve(ND * 2);
  _Float16* f1   = (_Float16*)carve(NF * 2);
  float*    x1   = (float*)carve(ND * 4);

  // weights fp32 -> fp16
  cvt_f32_to_f16<<<(int)(DD / 256), 256, 0, stream>>>(Wq, Wq16, (int)DD);
  cvt_f32_to_f16<<<(int)(DD / 256), 256, 0, stream>>>(Wk, Wk16, (int)DD);
  cvt_f32_to_f16<<<(int)(DD / 256), 256, 0, stream>>>(Wv, Wv16, (int)DD);
  cvt_f32_to_f16<<<(int)(DD / 256), 256, 0, stream>>>(Wo, Wo16, (int)DD);
  cvt_f32_to_f16<<<(int)(FD / 256), 256, 0, stream>>>(W1, W1h, (int)FD);
  cvt_f32_to_f16<<<(int)(FD / 256), 256, 0, stream>>>(W2, W2h, (int)FD);

  // --- attention half ---
  layernorm_to_f16<<<NTOK, 256, 0, stream>>>(x, g1, b1, hln);
  dim3 gqkv(DIM / BN, NTOK / BM);   // (8, 64)
  gemm_f16w<0><<<gqkv, 256, 0, stream>>>(hln, Wq16, bq, nullptr, nullptr, Qh,
                                         NTOK, DIM, DIM);
  gemm_f16w<0><<<gqkv, 256, 0, stream>>>(hln, Wk16, bk, nullptr, nullptr, Kh,
                                         NTOK, DIM, DIM);
  gemm_f16w<0><<<gqkv, 256, 0, stream>>>(hln, Wv16, bv, nullptr, nullptr, Vh,
                                         NTOK, DIM, DIM);
  attention_f16<<<BATCH * HEADS * (SEQ / 64), 128, 0, stream>>>(Qh, Kh, Vh, Ah);
  gemm_f16w<2><<<gqkv, 256, 0, stream>>>(Ah, Wo16, bo, x, x1, nullptr,
                                         NTOK, DIM, DIM);

  // --- FFN half ---
  layernorm_to_f16<<<NTOK, 256, 0, stream>>>(x1, g2, b2, h2);
  dim3 gffn1(FFN / BN, NTOK / BM);  // (32, 64)
  gemm_f16w<1><<<gffn1, 256, 0, stream>>>(h2, W1h, bf1, nullptr, nullptr, f1,
                                          NTOK, FFN, DIM);
  gemm_f16w<2><<<gqkv, 256, 0, stream>>>(f1, W2h, bf2, x1, out, nullptr,
                                         NTOK, DIM, FFN);
}
